// Conv2d_int8_STE_82617990906234
// MI455X (gfx1250) — 8-bit matrix operand order repaired and checked by reading, compiled, not run on hardware
//
#include <hip/hip_runtime.h>


#define BATCH  8
#define CIN    32
#define HH     32
#define WW     32
#define COUT   32
#define OH     32
#define OW     32
#define KDIM   288
#define KPAD   320

typedef __attribute__((ext_vector_type(8))) int v8i;

__global__ void init_amax_kernel(unsigned int* p) {
    if (threadIdx.x < 2) p[threadIdx.x] = 0u;
}

__global__ void amax_kernel(const float* __restrict__ in, int n,
                            unsigned int* __restrict__ out) {
    __shared__ float red[256];
    float m = 0.0f;
    for (int i = blockIdx.x * blockDim.x + threadIdx.x; i < n;
         i += gridDim.x * blockDim.x)
        m = fmaxf(m, fabsf(in[i]));
    red[threadIdx.x] = m;
    __syncthreads();
    for (int s = 128; s > 0; s >>= 1) {
        if ((int)threadIdx.x < s)
            red[threadIdx.x] = fmaxf(red[threadIdx.x], red[threadIdx.x + s]);
        __syncthreads();
    }
    if (threadIdx.x == 0) atomicMax(out, __float_as_uint(red[0]));
}

__global__ __launch_bounds__(128)
void conv_wmma_iu8_kernel(const float* __restrict__ x,
                          const float* __restrict__ w,
                          const float* __restrict__ bias,
                          const unsigned int* __restrict__ amax,
                          float* __restrict__ out) {
    __shared__ __align__(16) signed char lA[4][16][KPAD];
    __shared__ __align__(16) signed char lB[16][KPAD];

    const float ax = __uint_as_float(amax[0]);
    const float aw = __uint_as_float(amax[1]);
    const float sx = ax * (1.0f / 127.0f);
    const float sw = aw * (1.0f / 127.0f);
    const float inv_sx = 127.0f / ax;
    const float inv_sw = 127.0f / aw;

    const int tid  = threadIdx.x;
    const int wave = tid >> 5;
    const int lane = tid & 31;
    const int half = lane >> 4;
    const int l16  = lane & 15;

    const int tile_n  = blockIdx.y;
    const int tile_m  = blockIdx.x * 4 + wave;

    for (int idx = tid; idx < 16 * KPAD; idx += 128) {
        const int n = idx / KPAD;
        const int k = idx % KPAD;
        signed char v = 0;
        if (k < KDIM) {
            const int ci = k / 9;
            const int r  = k % 9;
            const float f = w[((tile_n * 16 + n) * CIN + ci) * 9 + r];
            float q = rintf(f * inv_sw);
            q = fminf(fmaxf(q, -128.0f), 127.0f);
            v = (signed char)(int)q;
        }
        lB[n][k] = v;
    }

    for (int idx = lane; idx < 16 * KPAD; idx += 32) {
        const int m = idx / KPAD;
        const int k = idx % KPAD;
        signed char v = 0;
        if (k < KDIM) {
            const int p  = tile_m * 16 + m;
            const int b  = p >> 10;
            const int oh = (p >> 5) & 31;
            const int ow = p & 31;
            const int ci = k / 9;
            const int r  = k % 9;
            const int kh = r / 3;
            const int kw = r % 3;
            const int ih = oh + kh - 1;
            const int iw = ow + kw - 1;
            if (ih >= 0 && ih < HH && iw >= 0 && iw < WW) {
                const float f = x[((b * CIN + ci) * HH + ih) * WW + iw];
                float q = rintf(f * inv_sx);
                q = fminf(fmaxf(q, -128.0f), 127.0f);
                v = (signed char)(int)q;
            }
        }
        lA[wave][m][k] = v;
    }

    __syncthreads();

    v8i acc = {0, 0, 0, 0, 0, 0, 0, 0};
#pragma unroll
    for (int kc = 0; kc < 5; ++kc) {
        const int K0 = kc * 64;
        v8i a, b;
#pragma unroll
        for (int v = 0; v < 8; ++v) {
            const int ka = K0 + ((v >> 1) << 4) + ((v & 1) << 2) + (half << 3);
            a[v] = *(const int*)&lA[wave][l16][ka];
            const int kb = K0 + ((v >> 1) << 4) + ((v & 1) << 2) + (half << 3);
            b[v] = *(const int*)&lB[l16][kb];
        }
        acc = __builtin_amdgcn_wmma_i32_16x16x64_iu8(
             true, a,  true, b, acc,
             false,  false);
    }

    const float scale = sx * sw;
    const int co = tile_n * 16 + l16;
    const float bi = bias[co];
#pragma unroll
    for (int r = 0; r < 8; ++r) {
        const int M  = tile_m * 16 + r + (half << 3);
        const int b  = M >> 10;
        const int oh = (M >> 5) & 31;
        const int ow = M & 31;
        out[((b * COUT + co) * OH + oh) * OW + ow] =
            (float)acc[r] * scale + bi;
    }
}

extern "C" void kernel_launch(void* const* d_in, const int* in_sizes, int n_in,
                              void* d_out, int out_size, void* d_ws, size_t ws_size,
                              hipStream_t stream) {
    const float* x    = (const float*)d_in[0];
    const float* w    = (const float*)d_in[1];
    const float* bias = (const float*)d_in[2];
    float* out = (float*)d_out;

    unsigned int* amax = (unsigned int*)d_ws;

    init_amax_kernel<<<1, 32, 0, stream>>>(amax);

    const int nx = in_sizes[0];
    const int nw = in_sizes[1];
    amax_kernel<<<128, 256, 0, stream>>>(x, nx, amax + 0);
    amax_kernel<<<16, 256, 0, stream>>>(w, nw, amax + 1);

    dim3 grid((BATCH * OH * OW) / 64, COUT / 16);
    conv_wmma_iu8_kernel<<<grid, 128, 0, stream>>>(x, w, bias, amax, out);
    (void)out_size; (void)ws_size; (void)n_in;
}
